// LocalMamba2D_77163382440856
// MI455X (gfx1250) — matrix operand order repaired and checked by reading, not compiled, not run on hardware
//
#include <hip/hip_runtime.h>
#include <hip/hip_bf16.h>
#include <stdint.h>

// ---------------------------------------------------------------------------
// LocalMamba2D block for MI455X (gfx1250, wave32, WMMA).
// Geometry: B=16, C=96, H=W=96 -> 4 quadrants of 48x48 (L=2304 pixels each).
// K=4 scan directions, N=8 state, R=6, Dproj=R+2N=22, CR=12.
// ---------------------------------------------------------------------------

#define BQ   16
#define CC   96
#define HF   96
#define WF   96
#define HS   48
#define LQ   2304          // 48*48
#define KD   4
#define NST  8
#define RR   6
#define DPJ  22            // R + 2N
#define CRR  12
#define NTOT ((size_t)BQ * CC * HF * WF)   // 14,155,776 elements (full tensor)

typedef __attribute__((ext_vector_type(16))) __bf16 v16bf;
typedef __attribute__((ext_vector_type(8)))  float  v8f;
typedef __attribute__((ext_vector_type(4)))  int    vi4;

#if defined(__gfx1250__) && __has_builtin(__builtin_amdgcn_global_load_async_to_lds_b128)
#define HAVE_ASYNC_LDS 1
// Builtin expects: (int4 addrspace(1)* gsrc, int4 addrspace(3)* ldst, imm offset, imm cpol)
#define ASYNC_B128(gsrc, ldst)                                              \
  __builtin_amdgcn_global_load_async_to_lds_b128(                           \
      (vi4 __attribute__((address_space(1)))*)(gsrc),                       \
      (vi4 __attribute__((address_space(3)))*)(ldst), 0, 0)
#endif

// Fast transcendentals: v_rcp_f32 instead of the IEEE div expansion.
__device__ __forceinline__ float fast_rcp(float x)  { return __builtin_amdgcn_rcpf(x); }
__device__ __forceinline__ float sigmoidf_(float v) { return fast_rcp(1.f + __expf(-v)); }
__device__ __forceinline__ float siluf(float v)     { return v * sigmoidf_(v); }
__device__ __forceinline__ float softplusf_(float v){ return (v > 20.f) ? v : log1pf(__expf(v)); }

__device__ __forceinline__ float wave_sum32(float v) {
#pragma unroll
  for (int off = 16; off; off >>= 1) v += __shfl_xor(v, off, 32);
  return v;
}

// Scan-direction pixel mapping within scan-space image (48x48):
// k=0: forward raster, k=1: column-major, k=2: reversed raster, k=3: reversed column-major.
__device__ __forceinline__ int scan_pix(int k, int l) {
  switch (k) {
    case 0:  return l;
    case 1:  { int h = l % HS, w = l / HS; return h * HS + w; }
    case 2:  return (LQ - 1) - l;
    default: { int m = (LQ - 1) - l; int h = m % HS, w = m / HS; return h * HS + w; }
  }
}

// ---------------------------------------------------------------------------
// K1: channel LayerNorm over C=96 (NCHW strided gather), output h in bf16,
//     pixel-major [q][b][p][c] so it is directly WMMA-A friendly.
//     One wave per pixel, 3 channels per lane.
// ---------------------------------------------------------------------------
__global__ void ln_h_kernel(const float* __restrict__ x,
                            const float* __restrict__ g,
                            const float* __restrict__ bt,
                            __bf16* __restrict__ hbuf) {
  const int tid  = threadIdx.x;
  const int wave = tid >> 5, lane = tid & 31;
  const int p = blockIdx.x * 4 + wave;
  const int b = blockIdx.y, q = blockIdx.z;
  const int hq = p / HS, wq = p % HS;
  const int h0 = (q >> 1) * HS, w0 = (q & 1) * HS;
  const size_t pixoff = (size_t)(h0 + hq) * WF + (w0 + wq);

  float vals[3];
  float s = 0.f, s2 = 0.f;
#pragma unroll
  for (int j = 0; j < 3; ++j) {
    const int c = lane + 32 * j;
    const float v = x[((size_t)b * CC + c) * (HF * WF) + pixoff];
    vals[j] = v; s += v; s2 += v * v;
  }
  s  = wave_sum32(s);
  s2 = wave_sum32(s2);
  const float mu  = s  * (1.f / CC);
  const float var = s2 * (1.f / CC) - mu * mu;
  const float rs  = rsqrtf(var + 1e-6f);

  const size_t base = ((size_t)(q * BQ + b) * LQ + p) * CC;
#pragma unroll
  for (int j = 0; j < 3; ++j) {
    const int c = lane + 32 * j;
    hbuf[base + c] = (__bf16)((vals[j] - mu) * rs * g[q * CC + c] + bt[q * CC + c]);
  }
}

// ---------------------------------------------------------------------------
// K2: dual conv1x1 GEMM with bf16 WMMA (v_wmma_f32_16x16x32_bf16).
//     Block = 256 threads (8 waves), each wave owns one 16-pixel M-tile.
//     Per wave: 2 weight sets x 6 N-tiles x 3 K-steps = 36 WMMAs.
//     H-tile staged with GLOBAL_LOAD_ASYNC_TO_LDS_B128 when available.
// ---------------------------------------------------------------------------
__global__ void gemm_dual_kernel(const __bf16* __restrict__ hbuf,
                                 const float* __restrict__ p1_w,
                                 const float* __restrict__ p1_b,
                                 const float* __restrict__ p2_w,
                                 const float* __restrict__ p2_b,
                                 float* __restrict__ abuf,
                                 float* __restrict__ z0buf) {
  __shared__ __bf16 Wlds[2][CC * CC];   // 36,864 B
  __shared__ __bf16 Hlds[128 * CC];     // 24,576 B  (total 61,440 B LDS)

  const int tid = threadIdx.x;
  const int b = blockIdx.y, q = blockIdx.z;
  const int p0 = blockIdx.x * 128;
  const int qoff = q * CC * CC;

  const size_t hbase = ((size_t)(q * BQ + b) * LQ + p0) * CC;

#ifdef HAVE_ASYNC_LDS
  {
    // 24,576 B = 1536 x b128 async transfers, ASYNCcnt-tracked, no VGPR hop.
    const uint4* src = (const uint4*)(hbuf + hbase);
    uint4* dst = (uint4*)Hlds;
    for (int i = tid; i < 128 * CC * 2 / 16; i += 256) {
      ASYNC_B128(src + i, dst + i);
    }
  }
#else
  {
    const uint32_t* src = (const uint32_t*)(hbuf + hbase);
    uint32_t* dst = (uint32_t*)Hlds;
    for (int i = tid; i < 128 * CC / 2; i += 256) dst[i] = src[i];
  }
#endif

  // Weights need fp32 -> bf16 conversion, so they go through VGPRs.
  for (int i = tid; i < CC * CC; i += 256) {
    Wlds[0][i] = (__bf16)p1_w[qoff + i];
    Wlds[1][i] = (__bf16)p2_w[qoff + i];
  }

#ifdef HAVE_ASYNC_LDS
  asm volatile("s_wait_asynccnt 0x0" ::: "memory");
#endif
  __syncthreads();

  const int wave = tid >> 5, lane = tid & 31;
  const int m  = lane & 15;     // A row within tile / B column / D column
  const int kh = lane >> 4;     // lane-half selector
  const int rowA = wave * 16 + m;

  // A fragments for the three K=32 steps (ISA: VGPR0-3 hold K={kh*8..kh*8+7},
  // VGPR4-7 hold K={16+kh*8..}).
  v16bf afrag[3];
#pragma unroll
  for (int ks = 0; ks < 3; ++ks) {
#pragma unroll
    for (int e = 0; e < 16; ++e) {
      const int k = ks * 32 + ((e < 8) ? (kh * 8 + e) : (16 + kh * 8 + (e - 8)));
      afrag[ks][e] = Hlds[rowA * CC + k];
    }
  }

#pragma unroll
  for (int wsel = 0; wsel < 2; ++wsel) {
    const float* bias = wsel ? p2_b : p1_b;
    float* outp = wsel ? z0buf : abuf;
    for (int nt = 0; nt < 6; ++nt) {
      v8f acc = {};
#pragma unroll
      for (int ks = 0; ks < 3; ++ks) {
        // B[k][n] = W[nt*16+n][k] (conv weight transposed); same K order as A:
        // elems 0-7 hold K={kh*8..kh*8+7}, elems 8-15 hold K={16+kh*8..}.
        v16bf bfrag;
#pragma unroll
        for (int e = 0; e < 16; ++e) {
          const int k = ks * 32 + ((e < 8) ? (kh * 8 + e) : (16 + kh * 8 + (e - 8)));
          bfrag[e] = Wlds[wsel][(nt * 16 + m) * CC + k];
        }
        acc = __builtin_amdgcn_wmma_f32_16x16x32_bf16(
            false, afrag[ks], false, bfrag, (short)0, acc, false, false);
      }
      const int ch = nt * 16 + m;
      const float bv = bias[q * CC + ch];
#pragma unroll
      for (int r = 0; r < 8; ++r) {
        const int pix = p0 + wave * 16 + r + 8 * kh;  // D: VGPR r -> row r + 8*laneHalf
        float v = acc[r] + bv;
        if (wsel == 0) v = siluf(v);
        outp[((size_t)(q * BQ + b) * LQ + pix) * CC + ch] = v;
      }
    }
  }
}

// ---------------------------------------------------------------------------
// K3: depthwise 3x3 + bias + SiLU, fused with the orientation transform into
//     scan space.  even block_idx: transpose HxW; odd: reverse both axes.
// ---------------------------------------------------------------------------
__global__ void dwconv_kernel(const float* __restrict__ z0,
                              const float* __restrict__ dw_w,
                              const float* __restrict__ dw_b,
                              const int* __restrict__ bidx,
                              float* __restrict__ zt) {
  const int c = threadIdx.x;
  const int p = blockIdx.x, b = blockIdx.y, q = blockIdx.z;
  const int hq = p / HS, wq = p % HS;
  const size_t qb = (size_t)(q * BQ + b);
  const float* zb = z0 + qb * LQ * CC;

  float acc = dw_b[q * CC + c];
#pragma unroll
  for (int ky = 0; ky < 3; ++ky) {
    const int hh = hq + ky - 1;
    if ((unsigned)hh >= HS) continue;
#pragma unroll
    for (int kx = 0; kx < 3; ++kx) {
      const int ww = wq + kx - 1;
      if ((unsigned)ww >= HS) continue;
      acc += zb[(size_t)(hh * HS + ww) * CC + c] *
             dw_w[((q * CC + c) * 3 + ky) * 3 + kx];
    }
  }
  const float v = siluf(acc);
  const bool even = ((bidx[0] & 1) == 0);
  const int pp = even ? (wq * HS + hq) : ((LQ - 1) - p);
  zt[(qb * LQ + pp) * CC + c] = v;
}

// ---------------------------------------------------------------------------
// K4: x_proj: x_dbl[b,k,l,d] = sum_c xproj_w[q,k,d,c] * xs_k[b,c,l].
//     xs_k is zt with the per-direction pixel permutation (never materialized).
// ---------------------------------------------------------------------------
__global__ void xdbl_kernel(const float* __restrict__ zt,
                            const float* __restrict__ xproj_w,
                            float* __restrict__ xdbl) {
  const int tid = threadIdx.x;
  const int d = tid & 31;
  const int sub = tid >> 5;                 // 0..7
  const int l = blockIdx.x * 8 + sub;
  const int bk = blockIdx.y;
  const int b = bk >> 2, k = bk & 3;
  const int q = blockIdx.z;
  if (d >= DPJ) return;

  const int p = scan_pix(k, l);
  const size_t qb = (size_t)(q * BQ + b);
  const float* zrow = zt + (qb * LQ + p) * CC;
  const float* wrow = xproj_w + ((size_t)(q * KD + k) * DPJ + d) * CC;

  float acc = 0.f;
#pragma unroll 8
  for (int c = 0; c < CC; ++c) acc += wrow[c] * zrow[c];

  xdbl[(((qb * KD + k) * LQ) + l) * DPJ + d] = acc;
}

// ---------------------------------------------------------------------------
// K5: selective scan.  One block per (q,b,k); thread c owns N=8 states in
//     registers.  delta recomputed on the fly (R=6 MACs).  x_dbl is staged
//     through LDS in 64-step chunks, double-buffered with async-to-LDS loads
//     so the next chunk's global latency hides behind the recurrence.
// ---------------------------------------------------------------------------
__global__ void scan_kernel(const float* __restrict__ zt,
                            const float* __restrict__ xdbl,
                            const float* __restrict__ A_logs,
                            const float* __restrict__ dtw_,
                            const float* __restrict__ dtb_,
                            const float* __restrict__ Ds_,
                            float* __restrict__ ycomb) {
#define CHW (64 * DPJ)                 // 1408 floats = 5632 B per chunk
  __shared__ float chunk[2][CHW];
  const int c = threadIdx.x;
  const int k = blockIdx.x, b = blockIdx.y, q = blockIdx.z;
  const int qk = q * KD + k;
  const size_t qb = (size_t)(q * BQ + b);

  float Areg[NST];
#pragma unroll
  for (int n = 0; n < NST; ++n)
    Areg[n] = -__expf(A_logs[((size_t)qk * CC + c) * NST + n]);
  float dtw[RR];
#pragma unroll
  for (int r = 0; r < RR; ++r) dtw[r] = dtw_[((size_t)qk * CC + c) * RR + r];
  const float dtb = dtb_[(size_t)qk * CC + c];
  const float Dsv = Ds_[(size_t)qk * CC + c];

  float hst[NST];
#pragma unroll
  for (int n = 0; n < NST; ++n) hst[n] = 0.f;

  const size_t xb = (qb * KD + k) * (size_t)LQ * DPJ;
  const float* ztb = zt + qb * LQ * CC;
  float* yb = ycomb + qb * LQ * CC;
  const int NCHUNK = LQ / 64;          // 36

#ifdef HAVE_ASYNC_LDS
  auto issue_chunk = [&](int ck, int buf) {
    const uint4* src = (const uint4*)(xdbl + xb + (size_t)ck * CHW);
    uint4* dst = (uint4*)chunk[buf];
    for (int i = c; i < CHW / 4; i += CC) {   // 352 x b128
      ASYNC_B128(src + i, dst + i);
    }
  };
  issue_chunk(0, 0);
  asm volatile("s_wait_asynccnt 0x0" ::: "memory");
  __syncthreads();
#else
  for (int i = c; i < CHW; i += CC) chunk[0][i] = xdbl[xb + i];
  __syncthreads();
#endif

  for (int ck = 0; ck < NCHUNK; ++ck) {
    const int buf = ck & 1;
#ifdef HAVE_ASYNC_LDS
    if (ck + 1 < NCHUNK) issue_chunk(ck + 1, buf ^ 1);   // overlap with compute
#else
    if (ck + 1 < NCHUNK) {
      // touch next chunk's cachelines (global_prefetch_b8); 96 threads x 64B
      // stride cover the 5.6 KB chunk.
      __builtin_prefetch(xdbl + xb + (size_t)(ck + 1) * CHW + (size_t)c * 16, 0, 0);
    }
#endif
    for (int t = 0; t < 64; ++t) {
      const float* xd = &chunk[buf][t * DPJ];
      float delta = dtb;
#pragma unroll
      for (int r = 0; r < RR; ++r) delta += dtw[r] * xd[r];
      delta = softplusf_(delta);

      const int l = ck * 64 + t;
      const int p = scan_pix(k, l);
      const float u = ztb[(size_t)p * CC + c];
      const float du = delta * u;
      float y = Dsv * u;
#pragma unroll
      for (int n = 0; n < NST; ++n) {
        const float dA = __expf(delta * Areg[n]);
        hst[n] = dA * hst[n] + du * xd[RR + n];
        y += hst[n] * xd[RR + NST + n];
      }
      atomicAdd(&yb[(size_t)p * CC + c], y);   // global_atomic_add_f32
    }
#ifdef HAVE_ASYNC_LDS
    asm volatile("s_wait_asynccnt 0x0" ::: "memory");
    __syncthreads();
#else
    __syncthreads();
    if (ck + 1 < NCHUNK) {
      for (int i = c; i < CHW; i += CC)
        chunk[buf ^ 1][i] = xdbl[xb + (size_t)(ck + 1) * CHW + i];
      __syncthreads();
    }
#endif
  }
#undef CHW
}

// ---------------------------------------------------------------------------
// K6: fused ssln-LN (scan space) -> inverse orientation -> p2n-LN -> gate
//     with `a`; accumulate per-(b,c) mean numerator via atomics.
// ---------------------------------------------------------------------------
__global__ void fuse_ln_gate_kernel(const float* __restrict__ ycomb,
                                    const float* __restrict__ abuf,
                                    const float* __restrict__ ssg,
                                    const float* __restrict__ ssb,
                                    const float* __restrict__ png,
                                    const float* __restrict__ pnb,
                                    const int* __restrict__ bidx,
                                    float* __restrict__ ybuf,
                                    float* __restrict__ sacc) {
  __shared__ float t1[CC], t2[CC];
  __shared__ float red[2];
  const int c = threadIdx.x;
  const int p = blockIdx.x, b = blockIdx.y, q = blockIdx.z;
  const size_t qb = (size_t)(q * BQ + b);

  const float v = ycomb[(qb * LQ + p) * CC + c];

  t1[c] = v; t2[c] = v * v;
  __syncthreads();
  if (c < 32) {
    float s1 = t1[c] + t1[c + 32] + t1[c + 64];
    float s2 = t2[c] + t2[c + 32] + t2[c + 64];
    s1 = wave_sum32(s1); s2 = wave_sum32(s2);
    if (c == 0) { red[0] = s1; red[1] = s2; }
  }
  __syncthreads();
  float mu  = red[0] * (1.f / CC);
  float var = red[1] * (1.f / CC) - mu * mu;
  float rs  = rsqrtf(var + 1e-6f);
  const float w1 = (v - mu) * rs * ssg[q * CC + c] + ssb[q * CC + c];
  __syncthreads();

  t1[c] = w1; t2[c] = w1 * w1;
  __syncthreads();
  if (c < 32) {
    float s1 = t1[c] + t1[c + 32] + t1[c + 64];
    float s2 = t2[c] + t2[c + 32] + t2[c + 64];
    s1 = wave_sum32(s1); s2 = wave_sum32(s2);
    if (c == 0) { red[0] = s1; red[1] = s2; }
  }
  __syncthreads();
  mu  = red[0] * (1.f / CC);
  var = red[1] * (1.f / CC) - mu * mu;
  rs  = rsqrtf(var + 1e-6f);
  const float w2 = (w1 - mu) * rs * png[q * CC + c] + pnb[q * CC + c];

  const bool even = ((bidx[0] & 1) == 0);
  const int pq = even ? ((p % HS) * HS + p / HS) : ((LQ - 1) - p);
  const float y = abuf[(qb * LQ + pq) * CC + c] * w2;
  ybuf[(qb * LQ + pq) * CC + c] = y;
  atomicAdd(&sacc[(q * BQ + b) * CC + c], y);
}

// ---------------------------------------------------------------------------
// K7: squeeze-excite channel attention (96 -> 12 -> 96) per (q,b).
// ---------------------------------------------------------------------------
__global__ void att_kernel(const float* __restrict__ sacc,
                           const float* __restrict__ w1,
                           const float* __restrict__ b1,
                           const float* __restrict__ w2,
                           const float* __restrict__ b2,
                           float* __restrict__ attb) {
  __shared__ float sm[CC];
  __shared__ float tv[CRR];
  const int c = threadIdx.x;
  const int b = blockIdx.x, q = blockIdx.y;
  const int qb = q * BQ + b;

  sm[c] = sacc[qb * CC + c] * (1.f / LQ);
  __syncthreads();
  if (c < CRR) {
    float t = b1[q * CRR + c];
    for (int cc = 0; cc < CC; ++cc) t += w1[(q * CRR + c) * CC + cc] * sm[cc];
    tv[c] = siluf(t);
  }
  __syncthreads();
  float a = b2[q * CC + c];
#pragma unroll
  for (int r = 0; r < CRR; ++r) a += w2[(q * CC + c) * CRR + r] * tv[r];
  attb[qb * CC + c] = sigmoidf_(a);
}

// ---------------------------------------------------------------------------
// K8: residual + quadrant reassembly into the full NCHW output.
// ---------------------------------------------------------------------------
__global__ void out_kernel(const float* __restrict__ x,
                           const float* __restrict__ ybuf,
                           const float* __restrict__ attb,
                           float* __restrict__ out) {
  const size_t idx = (size_t)blockIdx.x * 256 + threadIdx.x;
  if (idx >= NTOT) return;
  const int w = (int)(idx % WF);
  size_t t = idx / WF;
  const int hh = (int)(t % HF); t /= HF;
  const int c = (int)(t % CC);
  const int b = (int)(t / CC);
  const int q = ((hh >= HS) ? 2 : 0) + ((w >= HS) ? 1 : 0);
  const int hq = hh - ((hh >= HS) ? HS : 0);
  const int wq = w - ((w >= HS) ? HS : 0);
  const size_t qb = (size_t)(q * BQ + b);
  const float y = ybuf[(qb * LQ + hq * HS + wq) * CC + c];
  out[idx] = x[idx] + y * attb[qb * CC + c];
}

// ---------------------------------------------------------------------------
extern "C" void kernel_launch(void* const* d_in, const int* in_sizes, int n_in,
                              void* d_out, int out_size, void* d_ws, size_t ws_size,
                              hipStream_t stream) {
  const float* x        = (const float*)d_in[0];
  const float* ln_g     = (const float*)d_in[1];
  const float* ln_b     = (const float*)d_in[2];
  const float* p1_w     = (const float*)d_in[3];
  const float* p1_b     = (const float*)d_in[4];
  const float* p2in_w   = (const float*)d_in[5];
  const float* p2in_b   = (const float*)d_in[6];
  const float* dw_w     = (const float*)d_in[7];
  const float* dw_b     = (const float*)d_in[8];
  const float* xproj_w  = (const float*)d_in[9];
  const float* dtproj_w = (const float*)d_in[10];
  const float* dtproj_b = (const float*)d_in[11];
  const float* A_logs   = (const float*)d_in[12];
  const float* Ds       = (const float*)d_in[13];
  const float* ssln_g   = (const float*)d_in[14];
  const float* ssln_b   = (const float*)d_in[15];
  const float* p2n_g    = (const float*)d_in[16];
  const float* p2n_b    = (const float*)d_in[17];
  const float* ca_w1    = (const float*)d_in[18];
  const float* ca_b1    = (const float*)d_in[19];
  const float* ca_w2    = (const float*)d_in[20];
  const float* ca_b2    = (const float*)d_in[21];
  const int*   bidx     = (const int*)d_in[22];
  float* out = (float*)d_out;

  // Workspace carve-out (256B aligned):
  char* ws = (char*)d_ws;
  size_t off = 0;
  auto carve = [&](size_t bytes) -> void* {
    void* p = ws + off;
    off += (bytes + 255) & ~(size_t)255;
    return p;
  };
  __bf16* hbuf  = (__bf16*)carve(NTOT * 2);                         // 28.3 MB
  float*  abuf  = (float*)carve(NTOT * 4);                          // 56.6 MB
  float*  z0buf = (float*)carve(NTOT * 4);                          // 56.6 MB (reused as ybuf)
  float*  ztbuf = (float*)carve(NTOT * 4);                          // 56.6 MB
  float*  xdbl  = (float*)carve((size_t)4 * BQ * KD * LQ * DPJ * 4);// 51.9 MB
  float*  ycomb = (float*)carve(NTOT * 4);                          // 56.6 MB
  float*  sacc  = (float*)carve((size_t)4 * BQ * CC * 4);
  float*  attb  = (float*)carve((size_t)4 * BQ * CC * 4);

  (void)hipMemsetAsync(ycomb, 0, NTOT * 4, stream);
  (void)hipMemsetAsync(sacc, 0, (size_t)4 * BQ * CC * 4, stream);

  // 1) channel LN -> bf16 h (pixel-major)
  ln_h_kernel<<<dim3(LQ / 4, BQ, 4), 128, 0, stream>>>(x, ln_g, ln_b, hbuf);
  // 2) dual conv1x1 GEMM via bf16 WMMA -> a (SiLU'd) and z0
  gemm_dual_kernel<<<dim3(LQ / 128, BQ, 4), 256, 0, stream>>>(
      hbuf, p1_w, p1_b, p2in_w, p2in_b, abuf, z0buf);
  // 3) depthwise 3x3 + SiLU + orientation transform -> zt (scan space)
  dwconv_kernel<<<dim3(LQ, BQ, 4), CC, 0, stream>>>(z0buf, dw_w, dw_b, bidx, ztbuf);
  // 4) x_proj (22 outputs per (b,k,l))
  xdbl_kernel<<<dim3(LQ / 8, BQ * KD, 4), 256, 0, stream>>>(ztbuf, xproj_w, xdbl);
  // 5) selective scan + 4-direction combine (atomics into ycomb)
  scan_kernel<<<dim3(KD, BQ, 4), CC, 0, stream>>>(
      ztbuf, xdbl, A_logs, dtproj_w, dtproj_b, Ds, ycomb);
  // 6) ssln LN -> inverse orientation -> p2n LN -> gate; y into z0buf (dead)
  fuse_ln_gate_kernel<<<dim3(LQ, BQ, 4), CC, 0, stream>>>(
      ycomb, abuf, ssln_g, ssln_b, p2n_g, p2n_b, bidx, z0buf, sacc);
  // 7) channel attention
  att_kernel<<<dim3(BQ, 4), CC, 0, stream>>>(sacc, ca_w1, ca_b1, ca_w2, ca_b2, attb);
  // 8) residual + reassembly
  out_kernel<<<dim3((unsigned)((NTOT + 255) / 256)), 256, 0, stream>>>(
      x, z0buf, attb, out);
}
